// GatedSwitchesLayer_62861141344695
// MI455X (gfx1250) — matrix operand order repaired and checked by reading, not compiled, not run on hardware
//
#include <hip/hip_runtime.h>
#include <cstdint>
#include <cstddef>

// ---------------------------------------------------------------------------
// GatedSwitchesLayer for MI455X (gfx1250, wave32, WMMA + TDM).
// GEMMs: V_WMMA_F32_16X16X32_F16 with hi/lo split-precision f16 inputs
// (hi*hi + hi*lo + lo*hi) to approximate fp32 accuracy at matrix-pipe rates.
// Bulk global->LDS staging in K2 uses TENSOR_LOAD_TO_LDS (Tensor Data Mover).
// ---------------------------------------------------------------------------

typedef __attribute__((ext_vector_type(16))) _Float16 v16h;
typedef __attribute__((ext_vector_type(8)))  float    v8f;
typedef __attribute__((ext_vector_type(4)))  _Float16 h4;
typedef unsigned int u32x4 __attribute__((ext_vector_type(4)));
typedef int          i32x8 __attribute__((ext_vector_type(8)));
typedef int          i32x4 __attribute__((ext_vector_type(4)));

#define Bn 4
#define Vn 256
#define Hn 128

union Frag {
    v16h v;
    uint4 q[2];
};

__device__ __forceinline__ v8f wmma_f16(const v16h& a, const v16h& b, const v8f& c) {
    return __builtin_amdgcn_wmma_f32_16x16x32_f16(false, a, false, b, (short)0, c,
                                                  false, false);
}

__device__ __forceinline__ void split4(const float4 f, h4* hi, h4* lo) {
    h4 h_, l_;
    h_.x = (_Float16)f.x; l_.x = (_Float16)(f.x - (float)h_.x);
    h_.y = (_Float16)f.y; l_.y = (_Float16)(f.y - (float)h_.y);
    h_.z = (_Float16)f.z; l_.z = (_Float16)(f.z - (float)h_.z);
    h_.w = (_Float16)f.w; l_.w = (_Float16)(f.w - (float)h_.w);
    *hi = h_; *lo = l_;
}

__device__ __forceinline__ void split16(const float* __restrict__ p, v16h& hi, v16h& lo) {
#pragma unroll
    for (int t = 0; t < 16; ++t) {
        float x = p[t];
        _Float16 h_ = (_Float16)x;
        hi[t] = h_;
        lo[t] = (_Float16)(x - (float)h_);
    }
}

// ---------------------------------------------------------------------------
// Tensor Data Mover: 1-D copy of nElems fp32 from global to LDS.
// D# per CDNA5 ISA ch.8: group0 = count/lds_addr/global_addr/type,
// group1 = data_size=4B, tensor_dim0=tile_dim0=nElems, tensor_dim1=1,
// tile_dim1..4=0 (unused), tensor_dim0_stride=nElems. Groups 2/3 zero (<=2D).
// This toolchain exposes the 6-arg builtin (g0, g1, g2, g3, g4, cpol).
// ---------------------------------------------------------------------------
__device__ __forceinline__ void tdm_load_1d_f32(unsigned ldsOff, const void* gptr,
                                                unsigned nElems) {
    const unsigned long long ga = (unsigned long long)(uintptr_t)gptr;
    u32x4 g0;
    g0[0] = 1u;                                            // count=1, user mode
    g0[1] = ldsOff;                                        // lds_addr (bytes)
    g0[2] = (unsigned)(ga & 0xFFFFFFFFu);                  // global_addr[31:0]
    g0[3] = (unsigned)((ga >> 32) & 0x01FFFFFFu) | (2u << 30); // [56:32] | type=2
    i32x8 g1;
    g1[0] = (int)(2u << 16);                               // data_size = 4 bytes
    g1[1] = (int)((nElems & 0xFFFFu) << 16);               // tensor_dim0[15:0]
    g1[2] = (int)(((nElems >> 16) & 0xFFFFu) | (1u << 16));// tensor_dim0[31:16], tensor_dim1=1
    g1[3] = (int)((nElems & 0xFFFFu) << 16);               // tile_dim0
    g1[4] = 0;                                             // tile_dim1=0, tile_dim2=0
    g1[5] = (int)nElems;                                   // tensor_dim0_stride[31:0]
    g1[6] = 0;
    g1[7] = 0;
    const i32x4 z4 = {0, 0, 0, 0};
    const i32x8 z8 = {0, 0, 0, 0, 0, 0, 0, 0};
    __builtin_amdgcn_tensor_load_to_lds(g0, g1, z4, z4, z8, 0);
}

// ---------------------------------------------------------------------------
// K1: Uh = h@Uw.T + Ub ; Vh = h@Vw.T + Vb ; Ah = h@Aw.T + Ab
// ---------------------------------------------------------------------------
__global__ __launch_bounds__(256) void k1_linear3(
    const float* __restrict__ hIn,
    const float* __restrict__ Uw, const float* __restrict__ Ub,
    const float* __restrict__ Vw, const float* __restrict__ Vb,
    const float* __restrict__ Aw, const float* __restrict__ Ab,
    float* __restrict__ Uh, float* __restrict__ Vh, float* __restrict__ Ah)
{
    __shared__ _Float16 hhi[16 * Hn];
    __shared__ _Float16 hlo[16 * Hn];
    const int t  = threadIdx.x;
    const int g0 = blockIdx.x * 16;

    {   // stage 16x128 tile of h as split f16 (8 floats per thread)
        const float4* src = (const float4*)(hIn + (size_t)g0 * Hn + t * 8);
        float4 f0 = src[0], f1 = src[1];
        h4 hi, lo;
        split4(f0, &hi, &lo);
        *(h4*)(hhi + t * 8)     = hi;  *(h4*)(hlo + t * 8)     = lo;
        split4(f1, &hi, &lo);
        *(h4*)(hhi + t * 8 + 4) = hi;  *(h4*)(hlo + t * 8 + 4) = lo;
    }
    __syncthreads();

    const int lane  = t & 31;
    const int wave  = t >> 5;
    const int n     = lane & 15;
    const int abase = (lane >> 4) * 8;    // D-row offset
    const int kb    = (lane >> 4) * 16;   // B-frag k offset
    const int hcol  = wave * 16 + n;

    Frag ahi[4], alo[4];
#pragma unroll
    for (int c = 0; c < 4; ++c) {
        const _Float16* p = hhi + n * Hn + 32 * c + kb;
        ahi[c].q[0] = *(const uint4*)p;
        ahi[c].q[1] = *(const uint4*)(p + 8);
        const _Float16* q = hlo + n * Hn + 32 * c + kb;
        alo[c].q[0] = *(const uint4*)q;
        alo[c].q[1] = *(const uint4*)(q + 8);
    }

    const float* Ws[3] = {Uw, Vw, Aw};
    const float* bs[3] = {Ub, Vb, Ab};
    float*       Os[3] = {Uh, Vh, Ah};
#pragma unroll
    for (int x = 0; x < 3; ++x) {
        const float bias = bs[x][hcol];
        v8f acc;
#pragma unroll
        for (int v2 = 0; v2 < 8; ++v2) acc[v2] = bias;
#pragma unroll
        for (int c = 0; c < 4; ++c) {
            v16h bhi, blo;
            split16(Ws[x] + (size_t)hcol * Hn + 32 * c + kb, bhi, blo);
            acc = wmma_f16(ahi[c].v, bhi, acc);
            acc = wmma_f16(ahi[c].v, blo, acc);
            acc = wmma_f16(alo[c].v, bhi, acc);
        }
#pragma unroll
        for (int v2 = 0; v2 < 8; ++v2) {
            const int m = v2 + abase;
            Os[x][(size_t)(g0 + m) * Hn + hcol] = acc[v2];
        }
    }
}

// ---------------------------------------------------------------------------
// K2: per (b,i): Ce GEMM (256x128x128) + e_new + sigmoid-gated aggregation
// over j (complete inside the WG) + per-channel batchnorm partials.
// e slice staged raw via TDM (two 64KB chunks), then split LDS->LDS.
// LDS: 64KB raw + 128KB e hi/lo + 64KB Cw hi/lo + 3KB misc = 259KB.
// ---------------------------------------------------------------------------
__global__ __launch_bounds__(256) void k2_edge(
    const float* __restrict__ e,   const float* __restrict__ S,
    const float* __restrict__ Adj,
    const float* __restrict__ Cw,  const float* __restrict__ Cb,
    const float* __restrict__ AhW, const float* __restrict__ VhW,
    const float* __restrict__ UhW,
    float* __restrict__ enew, float* __restrict__ hnew,
    float* __restrict__ epsum, float* __restrict__ epsq)
{
    extern __shared__ char smem_raw[];
    float*    eraw = (float*)smem_raw;                    // 16384 f32 (64KB)
    _Float16* ehi  = (_Float16*)(smem_raw + 65536);       // 256*128 halves
    _Float16* elo  = ehi + Vn * Hn;
    _Float16* cwhi = elo + Vn * Hn;                       // 128*128 halves
    _Float16* cwlo = cwhi + Hn * Hn;
    float* s_row   = (float*)(cwlo + Hn * Hn);
    float* a_row   = s_row + Vn;
    float* ahirow  = a_row + Vn;
    float* cbl     = ahirow + Hn;

    const int t  = threadIdx.x;
    const int wg = blockIdx.x;
    const int b  = wg >> 8;
    const int i  = wg & 255;
    const size_t sliceBase = ((size_t)b * Vn + i) * Vn;   // row index into (B*V*V)
    const float* eSlice = e + sliceBase * Hn;
    const unsigned erawOff = (unsigned)(uintptr_t)(void*)eraw; // LDS byte offset

    // kick off TDM for e rows [0,128) while we stage everything else
    if (t < 32) {
        tdm_load_1d_f32(erawOff, eSlice, 128 * Hn);
    }

    {   // stage C_w: 64 contiguous elements per thread, split to f16 hi/lo
        const float4* src = (const float4*)(Cw + t * 64);
#pragma unroll 4
        for (int kk = 0; kk < 16; ++kk) {
            float4 f = src[kk];
            h4 hi, lo; split4(f, &hi, &lo);
            *(h4*)(cwhi + t * 64 + kk * 4) = hi;
            *(h4*)(cwlo + t * 64 + kk * 4) = lo;
        }
    }
    s_row[t] = S[sliceBase + t];
    a_row[t] = Adj[sliceBase + t];
    if (t < Hn) {
        ahirow[t] = AhW[((size_t)b * Vn + i) * Hn + t];
        cbl[t]    = Cb[t];
    }

    if (t < 32) { __builtin_amdgcn_s_wait_tensorcnt(0); }
    __syncthreads();
    {   // split-convert chunk 0 (rows 0..127): 64 elements per thread
        const float4* src = (const float4*)eraw + t * 16;
        h4* dhi = (h4*)(ehi + t * 64);
        h4* dlo = (h4*)(elo + t * 64);
#pragma unroll 4
        for (int kk = 0; kk < 16; ++kk) {
            float4 f = src[kk];
            h4 hi, lo; split4(f, &hi, &lo);
            dhi[kk] = hi;  dlo[kk] = lo;
        }
    }
    __syncthreads();                     // eraw free for reuse
    if (t < 32) {
        tdm_load_1d_f32(erawOff, eSlice + 128 * Hn, 128 * Hn);
        __builtin_amdgcn_s_wait_tensorcnt(0);
    }
    __syncthreads();
    {   // split-convert chunk 1 (rows 128..255)
        const float4* src = (const float4*)eraw + t * 16;
        h4* dhi = (h4*)(ehi + 128 * Hn + t * 64);
        h4* dlo = (h4*)(elo + 128 * Hn + t * 64);
#pragma unroll 4
        for (int kk = 0; kk < 16; ++kk) {
            float4 f = src[kk];
            h4 hi, lo; split4(f, &hi, &lo);
            dhi[kk] = hi;  dlo[kk] = lo;
        }
    }
    __syncthreads();

    const int lane  = t & 31;
    const int wave  = t >> 5;
    const int n     = lane & 15;
    const int abase = (lane >> 4) * 8;
    const int kb    = (lane >> 4) * 8;
    const int hcol  = wave * 16 + n;          // wave owns h in [16w, 16w+16)

    // C_w B-fragments: loaded once, held in registers for all 16 M tiles
    Frag bhi[4], blo[4];
#pragma unroll
    for (int c = 0; c < 4; ++c) {
        const _Float16* p = cwhi + hcol * Hn + 32 * c + kb;
        bhi[c].q[0] = *(const uint4*)p;
        bhi[c].q[1] = *(const uint4*)(p + 16);
        const _Float16* q = cwlo + hcol * Hn + 32 * c + kb;
        blo[c].q[0] = *(const uint4*)q;
        blo[c].q[1] = *(const uint4*)(q + 16);
    }
    const float ah_i = ahirow[hcol];
    const float cb_h = cbl[hcol];
    float aggl = 0.f, esl = 0.f, eql = 0.f;

    for (int mt = 0; mt < 16; ++mt) {
        const int j0 = mt * 16;
        v8f acc;
#pragma unroll
        for (int v2 = 0; v2 < 8; ++v2) {      // acc init = C_b + Ah_i + Ah_j
            const int j = j0 + v2 + abase;
            acc[v2] = cb_h + ah_i + AhW[((size_t)b * Vn + j) * Hn + hcol];
        }
#pragma unroll
        for (int c = 0; c < 4; ++c) {
            Frag fahi, falo;
            const _Float16* p = ehi + (j0 + n) * Hn + 32 * c + abase;
            fahi.q[0] = *(const uint4*)p;
            fahi.q[1] = *(const uint4*)(p + 16);
            const _Float16* q = elo + (j0 + n) * Hn + 32 * c + abase;
            falo.q[0] = *(const uint4*)q;
            falo.q[1] = *(const uint4*)(q + 16);
            acc = wmma_f16(fahi.v, bhi[c].v, acc);
            acc = wmma_f16(fahi.v, blo[c].v, acc);
            acc = wmma_f16(falo.v, bhi[c].v, acc);
        }
        // fused epilogue: scale by S, store e_new, gate, aggregate, stats
#pragma unroll
        for (int v2 = 0; v2 < 8; ++v2) {
            const int j = j0 + v2 + abase;
            const float val = s_row[j] * acc[v2];
            enew[(sliceBase + j) * Hn + hcol] = val;
            const float g  = 1.f / (1.f + __expf(-val));
            const float vh = VhW[((size_t)b * Vn + j) * Hn + hcol];
            aggl += (g + a_row[j]) * vh;
            esl  += val;
            eql  += val * val;
        }
    }
    // lanes L and L+16 hold complementary j rows of the same h column
    aggl += __shfl_xor(aggl, 16, 32);
    esl  += __shfl_xor(esl, 16, 32);
    eql  += __shfl_xor(eql, 16, 32);
    if (lane < 16) {
        const size_t ri = ((size_t)b * Vn + i) * Hn + hcol;
        hnew[ri] = UhW[ri] + aggl;
        epsum[(size_t)wg * Hn + hcol] = esl;
        epsq[(size_t)wg * Hn + hcol]  = eql;
    }
}

// ---------------------------------------------------------------------------
// K3: finalize batchnorm statistics -> per-channel scale/shift (deterministic
// fixed-order reduction over the 1024 per-WG partials).
// ---------------------------------------------------------------------------
__global__ void k3_stats(const float* __restrict__ hnew,
                         const float* __restrict__ epsum,
                         const float* __restrict__ epsq,
                         const float* __restrict__ gh, const float* __restrict__ bh,
                         const float* __restrict__ ge, const float* __restrict__ be,
                         float* __restrict__ scale_h, float* __restrict__ shift_h,
                         float* __restrict__ scale_e, float* __restrict__ shift_e)
{
    const int hq = threadIdx.x;   // 128 threads
    float hs = 0.f, hq2 = 0.f, es = 0.f, eq2 = 0.f;
    for (int r = 0; r < Bn * Vn; ++r) {
        const float v = hnew[r * Hn + hq];
        hs += v; hq2 += v * v;
        es += epsum[r * Hn + hq];
        eq2 += epsq[r * Hn + hq];
    }
    const float invNh = 1.f / (float)(Bn * Vn);
    const float mh    = hs * invNh;
    const float varh  = hq2 * invNh - mh * mh;
    const float sh    = gh[hq] * rsqrtf(varh + 1e-5f);
    scale_h[hq] = sh;
    shift_h[hq] = bh[hq] - mh * sh;

    const float invNe = 1.f / (float)((size_t)Bn * Vn * Vn);
    const float me    = es * invNe;
    const float vare  = eq2 * invNe - me * me;
    const float se    = ge[hq] * rsqrtf(vare + 1e-5f);
    scale_e[hq] = se;
    shift_e[hq] = be[hq] - me * se;
}

// ---------------------------------------------------------------------------
// K4 / K5: normalize + ReLU + residual streams.
// ---------------------------------------------------------------------------
__global__ __launch_bounds__(256) void k4_hout(
    const float* __restrict__ hnew, const float* __restrict__ scale_h,
    const float* __restrict__ shift_h, const float* __restrict__ hin,
    float* __restrict__ out)
{
    const int idx = blockIdx.x * 256 + threadIdx.x;
    const int hq  = idx & (Hn - 1);
    const float v = hnew[idx] * scale_h[hq] + shift_h[hq];
    out[idx] = hin[idx] + fmaxf(v, 0.f);
}

__global__ __launch_bounds__(256) void k5_eout(
    const float* __restrict__ enew, const float* __restrict__ scale_e,
    const float* __restrict__ shift_e, const float* __restrict__ ein,
    float* __restrict__ eout)
{
    const size_t idx = ((size_t)blockIdx.x * 256 + threadIdx.x) * 4;
    const int hq = (int)(idx & (Hn - 1));
    const float4 en = *(const float4*)(enew + idx);
    const float4 ei = *(const float4*)(ein + idx);
    const float4 sc = *(const float4*)(scale_e + hq);
    const float4 sf = *(const float4*)(shift_e + hq);
    float4 o;
    o.x = ei.x + fmaxf(en.x * sc.x + sf.x, 0.f);
    o.y = ei.y + fmaxf(en.y * sc.y + sf.y, 0.f);
    o.z = ei.z + fmaxf(en.z * sc.z + sf.z, 0.f);
    o.w = ei.w + fmaxf(en.w * sc.w + sf.w, 0.f);
    *(float4*)(eout + idx) = o;
}

// ---------------------------------------------------------------------------
extern "C" void kernel_launch(void* const* d_in, const int* in_sizes, int n_in,
                              void* d_out, int out_size, void* d_ws, size_t ws_size,
                              hipStream_t stream)
{
    (void)in_sizes; (void)n_in; (void)out_size; (void)ws_size;
    const float* h   = (const float*)d_in[0];
    const float* e   = (const float*)d_in[1];
    const float* Adj = (const float*)d_in[2];
    const float* S   = (const float*)d_in[3];
    const float* Uw  = (const float*)d_in[4];
    const float* Ub  = (const float*)d_in[5];
    const float* Vw  = (const float*)d_in[6];
    const float* Vb  = (const float*)d_in[7];
    const float* Aw  = (const float*)d_in[8];
    const float* Ab  = (const float*)d_in[9];
    const float* Cw  = (const float*)d_in[10];
    const float* Cb  = (const float*)d_in[11];
    const float* gh  = (const float*)d_in[12];
    const float* bh  = (const float*)d_in[13];
    const float* ge  = (const float*)d_in[14];
    const float* be  = (const float*)d_in[15];
    float* out = (float*)d_out;
    float* ws  = (float*)d_ws;

    const size_t NE = (size_t)Bn * Vn * Vn * Hn;   // 33,554,432
    const size_t NH = (size_t)Bn * Vn * Hn;        // 131,072

    float* enew    = ws;
    float* Uh      = enew + NE;
    float* Vh      = Uh + NH;
    float* Ah      = Vh + NH;
    float* hnew    = Ah + NH;
    float* epsum   = hnew + NH;
    float* epsq    = epsum + NH;
    float* scale_h = epsq + NH;
    float* shift_h = scale_h + Hn;
    float* scale_e = shift_h + Hn;
    float* shift_e = scale_e + Hn;

    k1_linear3<<<(Bn * Vn) / 16, 256, 0, stream>>>(h, Uw, Ub, Vw, Vb, Aw, Ab,
                                                   Uh, Vh, Ah);

    // 64KB raw TDM buffer + 128KB e hi/lo + 64KB Cw hi/lo + misc = 265,216 B
    const size_t smem2 = 65536
                       + (size_t)(Vn * Hn * 2 + Hn * Hn * 2) * sizeof(_Float16)
                       + (size_t)(Vn + Vn + Hn + Hn) * sizeof(float);
    k2_edge<<<Bn * Vn, 256, (int)smem2, stream>>>(e, S, Adj, Cw, Cb, Ah, Vh, Uh,
                                                  enew, hnew, epsum, epsq);

    k3_stats<<<1, Hn, 0, stream>>>(hnew, epsum, epsq, gh, bh, ge, be,
                                   scale_h, shift_h, scale_e, shift_e);

    k4_hout<<<(int)(NH / 256), 256, 0, stream>>>(hnew, scale_h, shift_h, h, out);
    k5_eout<<<(int)(NE / 4 / 256), 256, 0, stream>>>(enew, scale_e, shift_e, e,
                                                     out + NH);
}
